// MultiHeadAttention_20890720928261
// MI455X (gfx1250) — matrix operand order repaired and checked by reading, not compiled, not run on hardware
//
#include <hip/hip_runtime.h>
#include <hip/hip_bf16.h>
#include <stdint.h>
#include <math.h>

// ---------------------------------------------------------------------------
// MI455X (gfx1250) fused multi-head attention layer, bf16 WMMA everywhere.
// Async global->LDS staging (ASYNCcnt) for GEMM A-tiles.
// ---------------------------------------------------------------------------

typedef __attribute__((ext_vector_type(16))) __bf16 v16bf;
typedef __attribute__((ext_vector_type(8)))  float  v8f;

union FragBF { v16bf v; uint32_t u[8]; };

__device__ __forceinline__ v8f wmma_bf16_acc(const FragBF& a, const FragBF& b, v8f c) {
  // D(16x16,f32) = A(16x32,bf16) * B(32x16,bf16) + C
  return __builtin_amdgcn_wmma_f32_16x16x32_bf16(false, a.v, false, b.v,
                                                 (short)0, c, false, false);
}

// Generic pointer (known to be LDS) -> 32-bit LDS byte address for async DMA.
__device__ __forceinline__ uint32_t ldsOffset(const void* p) {
  return (uint32_t)(uintptr_t)(__attribute__((address_space(3))) const uint8_t*)p;
}

// Per-lane async copy: 16 bytes global -> LDS, tracked by ASYNCcnt.
__device__ __forceinline__ void asyncLoadB128(uint32_t lds_addr, const void* gaddr) {
  asm volatile("global_load_async_to_lds_b128 %0, %1, off"
               :: "v"(lds_addr), "v"((uint64_t)(uintptr_t)gaddr)
               : "memory");
}

// Problem sizes (fixed by the reference)
constexpr int Bn = 8, Sn = 1024, INn = 512, Dn = 1024, Hn = 16, DEPn = 64;
constexpr int MTOT = Bn * Sn;               // 8192 rows
constexpr int NPACK = 4 * Dn;               // 4096 packed cols (Q|K|V|res)

// Workspace layout (bytes)
constexpr size_t OFF_XB = 0;                                        // bf16 [8192][512]
constexpr size_t OFF_WP = OFF_XB + (size_t)MTOT * INn * 2;          // bf16 [512][4096]
constexpr size_t OFF_DB = OFF_WP + (size_t)INn * NPACK * 2;         // bf16 [1024][1024]
constexpr size_t OFF_QB = OFF_DB + (size_t)Dn * Dn * 2;             // bf16 [B*H][S][64]
constexpr size_t OFF_KB = OFF_QB + (size_t)Bn * Hn * Sn * DEPn * 2; // bf16 [B*H][S][64]
constexpr size_t OFF_VT = OFF_KB + (size_t)Bn * Hn * Sn * DEPn * 2; // bf16 [B*H][64][S]
constexpr size_t OFF_AO = OFF_VT + (size_t)Bn * Hn * Sn * DEPn * 2; // bf16 [8192][1024]
constexpr size_t OFF_Y  = OFF_AO + (size_t)MTOT * Dn * 2;           // f32  [8192][1024]

// ---------------------------------------------------------------------------
// Kernel 0: convert inputs/weights to bf16, pack the 4 input-projection
// weights into one [512][4096] matrix so QKV+residual is a single GEMM.
// ---------------------------------------------------------------------------
__global__ void mha_pack_convert(const float* __restrict__ x,
                                 const float* __restrict__ wq,
                                 const float* __restrict__ wk,
                                 const float* __restrict__ wv,
                                 const float* __restrict__ resw,
                                 const float* __restrict__ dw,
                                 __bf16* __restrict__ xb,
                                 __bf16* __restrict__ wp,
                                 __bf16* __restrict__ db) {
  const int tid = blockIdx.x * blockDim.x + threadIdx.x;
  const int nth = gridDim.x * blockDim.x;
  for (int i = tid; i < INn * NPACK; i += nth) {
    const int k = i >> 12, c = i & (NPACK - 1);
    float v;
    if (c < Dn)            v = wq[k * Dn + c];
    else if (c < 2 * Dn)   v = wk[k * Dn + (c - Dn)];
    else if (c < 3 * Dn)   v = wv[k * Dn + (c - 2 * Dn)];
    else                   v = resw[k * Dn + (c - 3 * Dn)];
    wp[i] = (__bf16)v;
  }
  for (int i = tid; i < Dn * Dn; i += nth) db[i] = (__bf16)dw[i];
  for (int i = tid; i < MTOT * INn; i += nth) xb[i] = (__bf16)x[i];
}

// ---------------------------------------------------------------------------
// LDS-staged bf16 WMMA GEMM: C[M][N] = A[M][K] * B[K][N], fp32 accumulate.
// Block tile 256x128, 8 waves (4x2), wave tile 64x64 (4x4 frags = 16 WMMAs
// per 32-wide K step).  A tile staged via global_load_async_to_lds_b128.
// EPI 0: QKV+residual scatter epilogue.  EPI 1: Y += acc + bias.
// ---------------------------------------------------------------------------
template <int EPI>
__launch_bounds__(256, 2)
__global__ void mha_gemm_wmma(const __bf16* __restrict__ A,
                              const __bf16* __restrict__ Bm,
                              int Ntot, int Ktot,
                              const float* __restrict__ bq,
                              const float* __restrict__ bk,
                              const float* __restrict__ bv,
                              const float* __restrict__ bd,
                              __bf16* __restrict__ Qb,
                              __bf16* __restrict__ Kb,
                              __bf16* __restrict__ Vt,
                              float* __restrict__ Y) {
  __shared__ uint32_t As[256][20];   // 256 rows x 32 bf16 (16 dwords) + pad->20
  __shared__ uint32_t Bt[128][20];   // 128 cols x 32 bf16 (16 dwords, K-major)

  const int t    = threadIdx.x;
  const int lane = t & 31;
  const int w    = t >> 5;        // 0..7
  const int wm   = w & 3;         // wave row quadrant (64 rows)
  const int wn   = w >> 2;        // wave col half (64 cols)
  const int half = lane >> 4;
  const int l16  = lane & 15;

  const int n0   = blockIdx.x * 128;
  const int row0 = blockIdx.y * 256;

  const uint32_t* Adw = (const uint32_t*)A;     // pairs of bf16
  const uint16_t* Bu  = (const uint16_t*)Bm;
  const int Kdw = Ktot >> 1;

  const uint32_t asLds = ldsOffset(&As[t][0]);  // this thread's A staging row

  v8f acc[4][4];
#pragma unroll
  for (int i = 0; i < 4; ++i)
#pragma unroll
    for (int j = 0; j < 4; ++j)
#pragma unroll
      for (int e = 0; e < 8; ++e) acc[i][j][e] = 0.0f;

  for (int kk = 0; kk < Ktot; kk += 32) {
    const int kdw0 = kk >> 1;
    // ---- cooperative tile loads -----------------------------------------
    {   // A: thread t stages full row (row0+t): 16 dwords via async DMA
      const uint8_t* gsrc = (const uint8_t*)(Adw + (size_t)(row0 + t) * Kdw + kdw0);
#pragma unroll
      for (int i = 0; i < 4; ++i)
        asyncLoadB128(asLds + 16u * i, gsrc + 16 * i);
    }
#pragma unroll
    for (int i = 0; i < 16; ++i) {                // B: 32x128, store transposed
      const int lin = t + 256 * i;
      const int kl = lin >> 7, nl = lin & 127;
      ((uint16_t*)Bt)[nl * 40 + kl] = Bu[(size_t)(kk + kl) * Ntot + n0 + nl];
    }
    if (kk + 32 < Ktot) {      // prefetch next K-step tiles into cache
      __builtin_prefetch(Adw + (size_t)(row0 + t) * Kdw + kdw0 + 16, 0, 1);
      __builtin_prefetch(Bu + (size_t)(kk + 32 + (t & 31)) * Ntot + n0 + ((t >> 5) << 4), 0, 1);
    }
    asm volatile("s_wait_asynccnt 0" ::: "memory");   // own async copies done
    __syncthreads();

    // ---- fragment gather + 16 WMMAs -------------------------------------
    FragBF a[4], b[4];
#pragma unroll
    for (int fm = 0; fm < 4; ++fm) {
      const uint32_t* rowp = &As[wm * 64 + fm * 16 + l16][0];
#pragma unroll
      for (int j = 0; j < 8; ++j)
        a[fm].u[j] = rowp[half * 4 + (j < 4 ? j : j + 4)];
    }
#pragma unroll
    for (int fn = 0; fn < 4; ++fn) {
      const uint32_t* rowp = &Bt[wn * 64 + fn * 16 + l16][0];
#pragma unroll
      for (int j = 0; j < 8; ++j)
        b[fn].u[j] = rowp[half * 4 + (j < 4 ? j : j + 4)];
    }
#pragma unroll
    for (int fm = 0; fm < 4; ++fm)
#pragma unroll
      for (int fn = 0; fn < 4; ++fn)
        acc[fm][fn] = wmma_bf16_acc(a[fm], b[fn], acc[fm][fn]);
    __syncthreads();
  }

  // ---- epilogue ----------------------------------------------------------
#pragma unroll
  for (int fm = 0; fm < 4; ++fm)
#pragma unroll
    for (int fn = 0; fn < 4; ++fn)
#pragma unroll
      for (int j = 0; j < 8; ++j) {
        const int r = row0 + wm * 64 + fm * 16 + half * 8 + j;
        const int c = n0 + wn * 64 + fn * 16 + l16;
        float v = acc[fm][fn][j];
        if (EPI == 0) {
          if (c < 3 * Dn) {
            const int seg = c >> 10;          // 0=Q 1=K 2=V
            const int cc = c & (Dn - 1);
            v += (seg == 0 ? bq : seg == 1 ? bk : bv)[cc];
            const int h = cc >> 6, d = cc & 63;
            const int bb = r >> 10, s = r & (Sn - 1);
            const size_t bh = (size_t)bb * Hn + h;
            if (seg == 0)      Qb[(bh * Sn + s) * DEPn + d] = (__bf16)v;
            else if (seg == 1) Kb[(bh * Sn + s) * DEPn + d] = (__bf16)v;
            else               Vt[(bh * DEPn + d) * Sn + s] = (__bf16)v;   // transposed
          } else {
            Y[(size_t)r * Dn + (c - 3 * Dn)] = v;                          // residual
          }
        } else {
          const size_t idx = (size_t)r * Dn + c;
          Y[idx] = Y[idx] + v + bd[c];
        }
      }
}

// ---------------------------------------------------------------------------
// Flash attention: one wave = 16 query rows, online softmax over 32-key tiles.
// grid = (S/128, B*H), block = 256 (8 waves).
// ---------------------------------------------------------------------------
__launch_bounds__(256, 2)
__global__ void mha_attn_wmma(const __bf16* __restrict__ Qb,
                              const __bf16* __restrict__ Kb,
                              const __bf16* __restrict__ Vt,
                              const int* __restrict__ mask,
                              __bf16* __restrict__ AO) {
  const int bh = blockIdx.y;
  const int bb = bh >> 4, h = bh & 15;
  const int t = threadIdx.x, lane = t & 31, w = t >> 5;
  const int half = lane >> 4, l16 = lane & 15;
  const int q0 = blockIdx.x * 128 + w * 16;

  const uint32_t* Qdw = (const uint32_t*)(Qb + (size_t)bh * Sn * DEPn); // [S][32]dw
  const uint32_t* Kdw = (const uint32_t*)(Kb + (size_t)bh * Sn * DEPn); // [S][32]dw
  const uint32_t* Vdw = (const uint32_t*)(Vt + (size_t)bh * DEPn * Sn); // [64][512]dw
  const int* mbase = mask + (size_t)h * Sn * Sn;

  // Q fragments (16 rows x 64 depth -> two 16x32 A-fragments), loaded once.
  FragBF qa[2];
  {
    const int qrow = q0 + l16;
#pragma unroll
    for (int ds = 0; ds < 2; ++ds)
#pragma unroll
      for (int j = 0; j < 8; ++j)
        qa[ds].u[j] = Qdw[(size_t)qrow * 32 + ds * 16 + half * 4 + (j < 4 ? j : j + 4)];
  }

  v8f o[4];
#pragma unroll
  for (int fn = 0; fn < 4; ++fn)
#pragma unroll
    for (int e = 0; e < 8; ++e) o[fn][e] = 0.0f;
  float m_run[8], l_run[8];
#pragma unroll
  for (int j = 0; j < 8; ++j) { m_run[j] = -INFINITY; l_run[j] = 0.0f; }
  const float scale = 0.125f;  // 1/sqrt(64)

  __shared__ uint32_t Pl[8][16][17];         // per-wave 16x32 bf16 P staging
  uint32_t (*pl)[17] = Pl[w];

  for (int kt = 0; kt < Sn; kt += 32) {
    // ---- scores: S(16x32) = Q(16x64) * K^T ------------------------------
    v8f s[2];
#pragma unroll
    for (int fn = 0; fn < 2; ++fn) {
      v8f c;
#pragma unroll
      for (int e = 0; e < 8; ++e) c[e] = 0.0f;
      const int krow = kt + fn * 16 + l16;
#pragma unroll
      for (int ds = 0; ds < 2; ++ds) {
        FragBF kb;
#pragma unroll
        for (int j = 0; j < 8; ++j)
          kb.u[j] = Kdw[(size_t)krow * 32 + ds * 16 + half * 4 + (j < 4 ? j : j + 4)];
        c = wmma_bf16_acc(qa[ds], kb, c);
      }
      s[fn] = c;
    }

    // ---- mask + scale + online softmax ----------------------------------
    float tmax[8];
#pragma unroll
    for (int j = 0; j < 8; ++j) {
      const int qg = q0 + half * 8 + j;
      const int* mr = mbase + (size_t)qg * Sn + kt;
      const float v0 = mr[l16]      ? s[0][j] * scale : -1e9f;
      const float v1 = mr[16 + l16] ? s[1][j] * scale : -1e9f;
      s[0][j] = v0; s[1][j] = v1;
      tmax[j] = fmaxf(v0, v1);
    }
#pragma unroll
    for (int off = 1; off < 16; off <<= 1)
#pragma unroll
      for (int j = 0; j < 8; ++j)
        tmax[j] = fmaxf(tmax[j], __shfl_xor(tmax[j], off, 16));

    float psum[8], fac[8];
#pragma unroll
    for (int j = 0; j < 8; ++j) {
      const float mn = fmaxf(m_run[j], tmax[j]);
      fac[j] = __expf(m_run[j] - mn);
      m_run[j] = mn;
      const float p0 = __expf(s[0][j] - mn);
      const float p1 = __expf(s[1][j] - mn);
      s[0][j] = p0; s[1][j] = p1;
      psum[j] = p0 + p1;
    }
#pragma unroll
    for (int off = 1; off < 16; off <<= 1)
#pragma unroll
      for (int j = 0; j < 8; ++j)
        psum[j] += __shfl_xor(psum[j], off, 16);
#pragma unroll
    for (int j = 0; j < 8; ++j) l_run[j] = l_run[j] * fac[j] + psum[j];
#pragma unroll
    for (int fn = 0; fn < 4; ++fn)
#pragma unroll
      for (int j = 0; j < 8; ++j) o[fn][j] *= fac[j];

    // ---- P: C-layout -> A-layout via per-wave LDS slab ------------------
    {
      __bf16* pb = (__bf16*)pl;                 // row stride 34 bf16
#pragma unroll
      for (int j = 0; j < 8; ++j) {
        const int r = half * 8 + j;
        pb[r * 34 + l16]      = (__bf16)s[0][j];
        pb[r * 34 + 16 + l16] = (__bf16)s[1][j];
      }
    }
    asm volatile("s_wait_dscnt 0" ::: "memory");   // same-wave LDS store->load
    FragBF pa;
    {
      const uint32_t* rowp = &pl[l16][0];
#pragma unroll
      for (int j = 0; j < 8; ++j)
        pa.u[j] = rowp[half * 4 + (j < 4 ? j : j + 4)];
    }

    // ---- O(16x64) += P(16x32) * V(32x64) --------------------------------
#pragma unroll
    for (int fn = 0; fn < 4; ++fn) {
      FragBF vb;
      const uint32_t* vrow = Vdw + (size_t)(fn * 16 + l16) * (Sn / 2);
#pragma unroll
      for (int j = 0; j < 8; ++j)
        vb.u[j] = vrow[(kt >> 1) + half * 4 + (j < 4 ? j : j + 4)];
      o[fn] = wmma_bf16_acc(pa, vb, o[fn]);
    }
  }

  // ---- finalize: divide by l, merge heads, store bf16 --------------------
#pragma unroll
  for (int j = 0; j < 8; ++j) l_run[j] = 1.0f / l_run[j];
#pragma unroll
  for (int fn = 0; fn < 4; ++fn)
#pragma unroll
    for (int j = 0; j < 8; ++j) {
      const int qg = q0 + half * 8 + j;
      const int col = h * DEPn + fn * 16 + l16;
      AO[((size_t)bb * Sn + qg) * Dn + col] = (__bf16)(o[fn][j] * l_run[j]);
    }
}

// ---------------------------------------------------------------------------
// LayerNorm over each 1024-wide row of Y, f32 output.
// ---------------------------------------------------------------------------
__launch_bounds__(256)
__global__ void mha_layernorm(const float* __restrict__ Y,
                              const float* __restrict__ g,
                              const float* __restrict__ b,
                              float* __restrict__ out) {
  const int row = blockIdx.x;
  const float* y = Y + (size_t)row * Dn;
  float s = 0.0f, s2 = 0.0f;
  for (int c = threadIdx.x; c < Dn; c += 256) {
    const float v = y[c];
    s += v; s2 += v * v;
  }
  __shared__ float rs[256], rs2[256];
  rs[threadIdx.x] = s; rs2[threadIdx.x] = s2;
  __syncthreads();
  for (int off = 128; off > 0; off >>= 1) {
    if (threadIdx.x < off) {
      rs[threadIdx.x]  += rs[threadIdx.x + off];
      rs2[threadIdx.x] += rs2[threadIdx.x + off];
    }
    __syncthreads();
  }
  const float mean = rs[0] * (1.0f / Dn);
  const float var  = rs2[0] * (1.0f / Dn) - mean * mean;
  const float inv  = rsqrtf(var + 1e-5f);
  for (int c = threadIdx.x; c < Dn; c += 256)
    out[(size_t)row * Dn + c] = (y[c] - mean) * inv * g[c] + b[c];
}

// ---------------------------------------------------------------------------
extern "C" void kernel_launch(void* const* d_in, const int* in_sizes, int n_in,
                              void* d_out, int out_size, void* d_ws, size_t ws_size,
                              hipStream_t stream) {
  const float* x     = (const float*)d_in[0];
  const int*   mask  = (const int*)d_in[1];
  const float* wq_w  = (const float*)d_in[2];
  const float* wq_b  = (const float*)d_in[3];
  const float* wk_w  = (const float*)d_in[4];
  const float* wk_b  = (const float*)d_in[5];
  const float* wv_w  = (const float*)d_in[6];
  const float* wv_b  = (const float*)d_in[7];
  const float* dns_w = (const float*)d_in[8];
  const float* dns_b = (const float*)d_in[9];
  const float* res_w = (const float*)d_in[10];
  const float* ln_g  = (const float*)d_in[11];
  const float* ln_b  = (const float*)d_in[12];

  char* ws = (char*)d_ws;
  __bf16* Xb = (__bf16*)(ws + OFF_XB);
  __bf16* Wp = (__bf16*)(ws + OFF_WP);
  __bf16* Db = (__bf16*)(ws + OFF_DB);
  __bf16* Qb = (__bf16*)(ws + OFF_QB);
  __bf16* Kb = (__bf16*)(ws + OFF_KB);
  __bf16* Vt = (__bf16*)(ws + OFF_VT);
  __bf16* AO = (__bf16*)(ws + OFF_AO);
  float*  Y  = (float*)(ws + OFF_Y);
  float*  out = (float*)d_out;

  // 0) convert + pack
  mha_pack_convert<<<1024, 256, 0, stream>>>(x, wq_w, wk_w, wv_w, res_w, dns_w,
                                             Xb, Wp, Db);
  // 1) QKV + residual: [8192,512] x [512,4096]
  mha_gemm_wmma<0><<<dim3(NPACK / 128, MTOT / 256), 256, 0, stream>>>(
      Xb, Wp, NPACK, INn, wq_b, wk_b, wv_b, nullptr, Qb, Kb, Vt, Y);
  // 2) attention
  mha_attn_wmma<<<dim3(Sn / 128, Bn * Hn), 256, 0, stream>>>(Qb, Kb, Vt, mask, AO);
  // 3) dense: [8192,1024] x [1024,1024], Y += acc + bias (residual already in Y)
  mha_gemm_wmma<1><<<dim3(Dn / 128, MTOT / 256), 256, 0, stream>>>(
      AO, Db, Dn, Dn, nullptr, nullptr, nullptr, dns_b, Qb, Kb, Vt, Y);
  // 4) layernorm -> out
  mha_layernorm<<<MTOT, 256, 0, stream>>>(Y, ln_g, ln_b, out);
}
